// Attention_5299989643594
// MI455X (gfx1250) — matrix operand order repaired and checked by reading, not compiled, not run on hardware
//
#include <hip/hip_runtime.h>
#include <hip/hip_bf16.h>

typedef _Float16 v8h  __attribute__((ext_vector_type(8)));
typedef _Float16 v16h __attribute__((ext_vector_type(16)));
typedef float    v8f  __attribute__((ext_vector_type(8)));

#define DIMC   2048
#define NH     16
#define QLORA  1536
#define KVLORA 512
#define QKN    128
#define QKR    64
#define QKH    192          // 128 + 64
#define VH     128
#define BB     2
#define SS     2048
#define SCALE_F 0.07216878364870322f   // 192^-0.5

// ---------------- workspace layout (element counts, _Float16) -------------
constexpr size_t SZ_X    = (size_t)BB*SS*DIMC;
constexpr size_t SZ_WQA  = (size_t)QLORA*DIMC;
constexpr size_t SZ_WQB  = (size_t)NH*QKH*QLORA;
constexpr size_t SZ_WKVA = (size_t)(KVLORA+QKR)*DIMC;
constexpr size_t SZ_WKVB = (size_t)NH*(QKN+VH)*KVLORA;
constexpr size_t SZ_WO   = (size_t)DIMC*DIMC;
constexpr size_t SZ_QA   = (size_t)BB*SS*QLORA;
constexpr size_t SZ_QAN  = SZ_QA;
constexpr size_t SZ_QF   = (size_t)BB*SS*NH*QKH;
constexpr size_t SZ_KVA  = (size_t)BB*SS*(KVLORA+QKR);
constexpr size_t SZ_KVCN = (size_t)BB*SS*KVLORA;
constexpr size_t SZ_KVB  = (size_t)BB*SS*NH*(QKN+VH);
constexpr size_t SZ_QATT = (size_t)BB*NH*SS*QKH;
constexpr size_t SZ_KATT = SZ_QATT;
constexpr size_t SZ_VT   = (size_t)BB*NH*VH*SS;
constexpr size_t SZ_CTX  = (size_t)BB*SS*NH*VH;

constexpr size_t O_X    = 0;
constexpr size_t O_WQA  = O_X    + SZ_X;
constexpr size_t O_WQB  = O_WQA  + SZ_WQA;
constexpr size_t O_WKVA = O_WQB  + SZ_WQB;
constexpr size_t O_WKVB = O_WKVA + SZ_WKVA;
constexpr size_t O_WO   = O_WKVB + SZ_WKVB;
constexpr size_t O_QA   = O_WO   + SZ_WO;
constexpr size_t O_QAN  = O_QA   + SZ_QA;
constexpr size_t O_QF   = O_QAN  + SZ_QAN;
constexpr size_t O_KVA  = O_QF   + SZ_QF;
constexpr size_t O_KVCN = O_KVA  + SZ_KVA;
constexpr size_t O_KVB  = O_KVCN + SZ_KVCN;
constexpr size_t O_QATT = O_KVB  + SZ_KVB;
constexpr size_t O_KATT = O_QATT + SZ_QATT;
constexpr size_t O_VT   = O_KATT + SZ_KATT;
constexpr size_t O_CTX  = O_VT   + SZ_VT;

// ---------------------------------------------------------------------------
__global__ void cvt_f32_to_f16(const float* __restrict__ in,
                               _Float16* __restrict__ out, size_t n) {
    size_t i = (size_t)blockIdx.x * blockDim.x + threadIdx.x;
    size_t stride = (size_t)gridDim.x * blockDim.x;
    for (; i < n; i += stride) out[i] = (_Float16)in[i];
}

// ---------------------------------------------------------------------------
__device__ __forceinline__ v16h pack16(v8h lo, v8h hi) {
    return __builtin_shufflevector(lo, hi, 0,1,2,3,4,5,6,7,8,9,10,11,12,13,14,15);
}

// C(M,N) = A(M,K) * W(N,K)^T.  One wave computes a 64x64 tile with a 4x4
// grid of WMMA accumulators: 16 b128 loads feed 16 v_wmma per k-chunk (4x the
// operand reuse of a naive 16x16 tile).  Needs ~230 VGPRs -> compiled at
// occupancy 1 (launch_bounds min-waves-per-EU = 1) so nothing spills; latency
// is hidden by the 16 independent accumulator chains, not by wave count.
// Operand register layouts follow CDNA5 ISA 7.12.2: A lanes 0-15 hold
// K{0..7,16..23}, lanes 16-31 hold K{8..15,24..31}; B lane = column, lane
// halves hold the same K split as A.
template <typename OUT_T>
__device__ __forceinline__ void gemm_tile64(const _Float16* __restrict__ A,
                                            const _Float16* __restrict__ W,
                                            OUT_T* __restrict__ C,
                                            int M, int N, int K) {
    int wave = (int)((blockIdx.x * blockDim.x + threadIdx.x) >> 5);
    int lane = threadIdx.x & 31;
    int n64 = N >> 6;
    int mt = wave / n64;
    int nt = wave - mt * n64;
    if (mt >= (M >> 6)) return;                 // uniform per wave

    int lane15 = lane & 15;
    int hiHalf = (lane & 16) ? 1 : 0;
    const _Float16* arow[4];
    const _Float16* wrow[4];
#pragma unroll
    for (int i = 0; i < 4; ++i) {
        arow[i] = A + (size_t)(mt * 64 + i * 16 + lane15) * K + (hiHalf ? 8 : 0);
        wrow[i] = W + (size_t)(nt * 64 + i * 16 + lane15) * K + (hiHalf ? 8 : 0);
    }
    v8f acc[16];
#pragma unroll
    for (int i = 0; i < 16; ++i) acc[i] = (v8f){0.f,0.f,0.f,0.f,0.f,0.f,0.f,0.f};

    for (int kb = 0; kb < K; kb += 32) {
        if ((kb & 255) == 0 && kb + 256 < K) {
#pragma unroll
            for (int i = 0; i < 4; ++i) {
                __builtin_prefetch(arow[i] + kb + 256, 0, 0);
                __builtin_prefetch(wrow[i] + kb + 256, 0, 0);
            }
        }
        v16h a[4], b[4];
#pragma unroll
        for (int i = 0; i < 4; ++i) {
            a[i] = pack16(*(const v8h*)(arow[i] + kb),
                          *(const v8h*)(arow[i] + kb + 16));
            b[i] = pack16(*(const v8h*)(wrow[i] + kb),
                          *(const v8h*)(wrow[i] + kb + 16));
        }
#pragma unroll
        for (int mi = 0; mi < 4; ++mi)
#pragma unroll
            for (int ni = 0; ni < 4; ++ni)
                acc[mi * 4 + ni] = __builtin_amdgcn_wmma_f32_16x16x32_f16(
                    false, a[mi], false, b[ni], (short)0, acc[mi * 4 + ni],
                    false, false);
    }
#pragma unroll
    for (int mi = 0; mi < 4; ++mi)
#pragma unroll
        for (int ni = 0; ni < 4; ++ni)
#pragma unroll
            for (int r = 0; r < 8; ++r) {
                int row = mt * 64 + mi * 16 + (hiHalf ? 8 + r : r);
                C[(size_t)row * N + nt * 64 + ni * 16 + lane15] =
                    (OUT_T)acc[mi * 4 + ni][r];
            }
}

__global__ __launch_bounds__(256, 1)
void gemm_wmma_h(const _Float16* __restrict__ A,
                 const _Float16* __restrict__ W,
                 _Float16* __restrict__ C, int M, int N, int K) {
    gemm_tile64<_Float16>(A, W, C, M, N, K);
}
__global__ __launch_bounds__(256, 1)
void gemm_wmma_f(const _Float16* __restrict__ A,
                 const _Float16* __restrict__ W,
                 float* __restrict__ C, int M, int N, int K) {
    gemm_tile64<float>(A, W, C, M, N, K);
}

// ---------------------------------------------------------------------------
// RMSNorm: one block per row (f16 in, f16 out, f32 weight).
__global__ void rmsnorm_kernel(const _Float16* __restrict__ in,
                               const float* __restrict__ w,
                               _Float16* __restrict__ out,
                               int len, int in_stride) {
    int row = blockIdx.x;
    const _Float16* r = in + (size_t)row * in_stride;
    __shared__ float red[256];
    float acc = 0.f;
    for (int i = threadIdx.x; i < len; i += 256) {
        float v = (float)r[i];
        acc += v * v;
    }
    red[threadIdx.x] = acc;
    __syncthreads();
    for (int off = 128; off > 0; off >>= 1) {
        if ((int)threadIdx.x < off) red[threadIdx.x] += red[threadIdx.x + off];
        __syncthreads();
    }
    float inv = rsqrtf(red[0] / (float)len + 1e-6f);
    for (int i = threadIdx.x; i < len; i += 256)
        out[(size_t)row * len + i] = (_Float16)((float)r[i] * inv * w[i]);
}

// ---------------------------------------------------------------------------
// RoPE on q + repack [b][s][h][192] -> [b][h][s][192]
__global__ void rope_q_kernel(const _Float16* __restrict__ qf,
                              const float* __restrict__ fcos,
                              const float* __restrict__ fsin,
                              _Float16* __restrict__ qatt) {
    int idx = blockIdx.x;                 // b*S*H
    int h = idx % NH;
    int s = (idx / NH) % SS;
    int b = idx / (NH * SS);
    const _Float16* src = qf + ((size_t)(b * SS + s) * (NH * QKH)) + (size_t)h * QKH;
    _Float16* dst = qatt + (((size_t)(b * NH + h) * SS + s) * QKH);
    int t = threadIdx.x;
    if (t < 64) {                          // copy nope part (128 elems)
        dst[2 * t]     = src[2 * t];
        dst[2 * t + 1] = src[2 * t + 1];
    } else if (t < 96) {                   // rope part: 32 pairs
        int p = t - 64;
        float xr = (float)src[QKN + 2 * p];
        float xi = (float)src[QKN + 2 * p + 1];
        float c = fcos[(size_t)s * 32 + p];
        float sn = fsin[(size_t)s * 32 + p];
        dst[QKN + 2 * p]     = (_Float16)(xr * c - xi * sn);
        dst[QKN + 2 * p + 1] = (_Float16)(xr * sn + xi * c);
    }
}

// ---------------------------------------------------------------------------
// Build K rows [b][h][s][192] = k_nope || rope(k_pe) (broadcast), and
// transposed V [b][h][128][S].
__global__ void kv_pack_kernel(const _Float16* __restrict__ kvb,   // [b][s][h*256]
                               const _Float16* __restrict__ kva,   // [b][s][576]
                               const float* __restrict__ fcos,
                               const float* __restrict__ fsin,
                               _Float16* __restrict__ katt,
                               _Float16* __restrict__ vT) {
    int s = blockIdx.x % SS;
    int b = blockIdx.x / SS;
    __shared__ _Float16 kpe[QKR];
    int t = threadIdx.x;
    if (t < 32) {
        const _Float16* src = kva + (size_t)(b * SS + s) * (KVLORA + QKR) + KVLORA;
        float xr = (float)src[2 * t];
        float xi = (float)src[2 * t + 1];
        float c = fcos[(size_t)s * 32 + t];
        float sn = fsin[(size_t)s * 32 + t];
        kpe[2 * t]     = (_Float16)(xr * c - xi * sn);
        kpe[2 * t + 1] = (_Float16)(xr * sn + xi * c);
    }
    __syncthreads();
    const _Float16* row = kvb + (size_t)(b * SS + s) * (NH * (QKN + VH));
    for (int idx = t; idx < NH * QKH; idx += 256) {
        int h = idx / QKH, i = idx - h * QKH;
        _Float16 v = (i < QKN) ? row[(size_t)h * (QKN + VH) + i] : kpe[i - QKN];
        katt[(((size_t)(b * NH + h) * SS + s) * QKH) + i] = v;
    }
    for (int idx = t; idx < NH * VH; idx += 256) {
        int h = idx / VH, d = idx - h * VH;
        vT[((size_t)(b * NH + h) * VH + d) * SS + s] = row[(size_t)h * (QKN + VH) + QKN + d];
    }
}

// ---------------------------------------------------------------------------
// Flash attention: one wave per (b, h, 16-query tile); 32 keys per iteration
// (two 16x16 score tiles -> one full-K P operand for PV).  ~160 VGPRs live,
// so compile at occupancy 1 to avoid spills.
__global__ __launch_bounds__(32, 1)
void attn_kernel(const _Float16* __restrict__ q,    // [b][h][S][192]
                 const _Float16* __restrict__ k,    // [b][h][S][192]
                 const _Float16* __restrict__ vT,   // [b][h][128][S]
                 _Float16* __restrict__ ctx) {      // [b][s][h*128]
    int tiles = SS / 16;
    int qt = blockIdx.x % tiles;
    int h = (blockIdx.x / tiles) % NH;
    int b = blockIdx.x / (tiles * NH);
    int lane = threadIdx.x & 31;
    int lane15 = lane & 15;
    int hiHalf = (lane & 16) ? 1 : 0;
    int qbase = qt * 16;

    // Q tile -> 6 A-operands (K-chunks of 32 over head dim 192)
    const _Float16* qrow = q + (((size_t)(b * NH + h) * SS + qbase + lane15) * QKH)
                             + (hiHalf ? 8 : 0);
    v16h qa[6];
#pragma unroll
    for (int c = 0; c < 6; ++c) {
        v8h lo = *(const v8h*)(qrow + c * 32);
        v8h hi = *(const v8h*)(qrow + c * 32 + 16);
        qa[c] = pack16(lo, hi);
    }

    v8f acc[8];
    float mrow[8], lrow[8];
#pragma unroll
    for (int n = 0; n < 8; ++n) acc[n] = (v8f){0.f,0.f,0.f,0.f,0.f,0.f,0.f,0.f};
#pragma unroll
    for (int r = 0; r < 8; ++r) { mrow[r] = -3.0e38f; lrow[r] = 0.f; }

    __shared__ _Float16 pbuf[16 * 32];

    const _Float16* Kb = k + ((size_t)(b * NH + h) * SS) * QKH;
    const _Float16* Vb = vT + ((size_t)(b * NH + h) * VH) * SS;

    int jend = qbase + 16;   // causal extent (exclusive)
    for (int jb = 0; jb < jend; jb += 32) {
        bool have2 = (jb + 16) < jend;       // uniform across the wave
        // ---- scores for keys jb..jb+15 and (if present) jb+16..jb+31 ----
        v8f s0 = (v8f){0.f,0.f,0.f,0.f,0.f,0.f,0.f,0.f};
        v8f s1 = (v8f){0.f,0.f,0.f,0.f,0.f,0.f,0.f,0.f};
        const _Float16* krow0 = Kb + (size_t)(jb + lane15) * QKH + (hiHalf ? 8 : 0);
        const _Float16* krow1 = krow0 + (have2 ? (size_t)16 * QKH : 0);
#pragma unroll
        for (int c = 0; c < 6; ++c) {
            v16h kb0 = pack16(*(const v8h*)(krow0 + c * 32),
                              *(const v8h*)(krow0 + c * 32 + 16));
            s0 = __builtin_amdgcn_wmma_f32_16x16x32_f16(false, qa[c], false, kb0,
                                                        (short)0, s0, false, false);
        }
        if (have2) {
#pragma unroll
            for (int c = 0; c < 6; ++c) {
                v16h kb1 = pack16(*(const v8h*)(krow1 + c * 32),
                                  *(const v8h*)(krow1 + c * 32 + 16));
                s1 = __builtin_amdgcn_wmma_f32_16x16x32_f16(false, qa[c], false, kb1,
                                                            (short)0, s1, false, false);
            }
        }
        // ---- causal mask + online softmax over 32 columns ----
        float alpha[8];
        int col0 = jb + lane15;
        int col1 = jb + 16 + lane15;
#pragma unroll
        for (int r = 0; r < 8; ++r) {
            int rowg = qbase + (hiHalf ? 8 + r : r);
            float v0 = s0[r] * SCALE_F;
            if (col0 > rowg) v0 = -3.0e38f;
            float v1 = have2 ? s1[r] * SCALE_F : -3.0e38f;
            if (col1 > rowg) v1 = -3.0e38f;
            float mx = fmaxf(v0, v1);
            mx = fmaxf(mx, __shfl_xor(mx, 1, 32));
            mx = fmaxf(mx, __shfl_xor(mx, 2, 32));
            mx = fmaxf(mx, __shfl_xor(mx, 4, 32));
            mx = fmaxf(mx, __shfl_xor(mx, 8, 32));
            float mnew = fmaxf(mrow[r], mx);
            float p0 = __expf(v0 - mnew);
            float p1 = __expf(v1 - mnew);
            float ps = p0 + p1;
            ps += __shfl_xor(ps, 1, 32);
            ps += __shfl_xor(ps, 2, 32);
            ps += __shfl_xor(ps, 4, 32);
            ps += __shfl_xor(ps, 8, 32);
            alpha[r] = __expf(mrow[r] - mnew);
            lrow[r] = lrow[r] * alpha[r] + ps;
            mrow[r] = mnew;
            int prow = (hiHalf ? 8 : 0) + r;
            pbuf[prow * 32 + lane15]      = (_Float16)p0;
            pbuf[prow * 32 + 16 + lane15] = (_Float16)p1;
        }
#pragma unroll
        for (int n = 0; n < 8; ++n)
#pragma unroll
            for (int r = 0; r < 8; ++r) acc[n][r] *= alpha[r];
        __syncthreads();
        // ---- P (16x32) as A-operand ----
        const _Float16* pb = pbuf + lane15 * 32 + (hiHalf ? 8 : 0);
        v16h pa = pack16(*(const v8h*)(pb), *(const v8h*)(pb + 16));
        // ---- PV: acc += P * V (8 chunks of 16 v-dims) ----
#pragma unroll
        for (int n = 0; n < 8; ++n) {
            // B operand: lanes 0-15 hold keys jb+{0..7,16..23}, lanes 16-31 keys
            // jb+{8..15,24..31} (keys jb+16.. when absent have zero P entries,
            // so the products vanish).
            const _Float16* vrow = Vb + (size_t)(n * 16 + lane15) * SS + jb
                                      + (hiHalf ? 8 : 0);
            v16h vb16 = pack16(*(const v8h*)(vrow), *(const v8h*)(vrow + 16));
            acc[n] = __builtin_amdgcn_wmma_f32_16x16x32_f16(false, pa, false, vb16,
                                                            (short)0, acc[n],
                                                            false, false);
        }
        __syncthreads();
    }
    // ---- epilogue ----
#pragma unroll
    for (int n = 0; n < 8; ++n) {
#pragma unroll
        for (int r = 0; r < 8; ++r) {
            int rowg = qbase + (hiHalf ? 8 + r : r);
            int d = n * 16 + lane15;
            float o = acc[n][r] / lrow[r];
            ctx[((size_t)b * SS + rowg) * (NH * VH) + (size_t)h * VH + d] = (_Float16)o;
        }
    }
}

// ---------------------------------------------------------------------------
extern "C" void kernel_launch(void* const* d_in, const int* in_sizes, int n_in,
                              void* d_out, int out_size, void* d_ws, size_t ws_size,
                              hipStream_t stream) {
    const float* x     = (const float*)d_in[0];
    const float* fcos  = (const float*)d_in[1];
    const float* fsin  = (const float*)d_in[2];
    const float* wq_a  = (const float*)d_in[3];
    const float* qnw   = (const float*)d_in[4];
    const float* wq_b  = (const float*)d_in[5];
    const float* wkv_a = (const float*)d_in[6];
    const float* kvnw  = (const float*)d_in[7];
    const float* wkv_b = (const float*)d_in[8];
    const float* wo    = (const float*)d_in[9];
    float* out = (float*)d_out;

    _Float16* ws = (_Float16*)d_ws;
    _Float16* hX    = ws + O_X;
    _Float16* hWQA  = ws + O_WQA;
    _Float16* hWQB  = ws + O_WQB;
    _Float16* hWKVA = ws + O_WKVA;
    _Float16* hWKVB = ws + O_WKVB;
    _Float16* hWO   = ws + O_WO;
    _Float16* hQA   = ws + O_QA;
    _Float16* hQAN  = ws + O_QAN;
    _Float16* hQF   = ws + O_QF;
    _Float16* hKVA  = ws + O_KVA;
    _Float16* hKVCN = ws + O_KVCN;
    _Float16* hKVB  = ws + O_KVB;
    _Float16* hQATT = ws + O_QATT;
    _Float16* hKATT = ws + O_KATT;
    _Float16* hVT   = ws + O_VT;
    _Float16* hCTX  = ws + O_CTX;

    auto cvt = [&](const float* src, _Float16* dst, size_t n) {
        int blocks = (int)((n + 255) / 256);
        if (blocks > 4096) blocks = 4096;
        cvt_f32_to_f16<<<blocks, 256, 0, stream>>>(src, dst, n);
    };
    cvt(x,     hX,    SZ_X);
    cvt(wq_a,  hWQA,  SZ_WQA);
    cvt(wq_b,  hWQB,  SZ_WQB);
    cvt(wkv_a, hWKVA, SZ_WKVA);
    cvt(wkv_b, hWKVB, SZ_WKVB);
    cvt(wo,    hWO,   SZ_WO);

    auto gemm_h = [&](const _Float16* A, const _Float16* W, _Float16* C,
                      int M, int N, int K) {
        int waves = (M / 64) * (N / 64);
        int blocks = (waves + 7) / 8;
        gemm_wmma_h<<<blocks, 256, 0, stream>>>(A, W, C, M, N, K);
    };
    const int MROWS = BB * SS;   // 4096

    // q_a = x @ wq_a.T
    gemm_h(hX, hWQA, hQA, MROWS, QLORA, DIMC);
    // rms_norm(q_a)
    rmsnorm_kernel<<<MROWS, 256, 0, stream>>>(hQA, qnw, hQAN, QLORA, QLORA);
    // q = q_a_n @ wq_b.T
    gemm_h(hQAN, hWQB, hQF, MROWS, NH * QKH, QLORA);
    // kv_a = x @ wkv_a.T
    gemm_h(hX, hWKVA, hKVA, MROWS, KVLORA + QKR, DIMC);
    // rms_norm(kv_c)
    rmsnorm_kernel<<<MROWS, 256, 0, stream>>>(hKVA, kvnw, hKVCN, KVLORA, KVLORA + QKR);
    // kv = kv_c_n @ wkv_b.T
    gemm_h(hKVCN, hWKVB, hKVB, MROWS, NH * (QKN + VH), KVLORA);

    // rope + repack
    rope_q_kernel<<<BB * SS * NH, 128, 0, stream>>>(hQF, fcos, fsin, hQATT);
    kv_pack_kernel<<<BB * SS, 256, 0, stream>>>(hKVB, hKVA, fcos, fsin, hKATT, hVT);

    // flash attention
    attn_kernel<<<BB * NH * (SS / 16), 32, 0, stream>>>(hQATT, hKATT, hVT, hCTX);

    // out = ctx @ wo.T  (fp32 output)
    {
        int waves = (MROWS / 64) * (DIMC / 64);
        int blocks = (waves + 7) / 8;
        gemm_wmma_f<<<blocks, 256, 0, stream>>>(hCTX, hWO, out, MROWS, DIMC, DIMC);
    }
}
